// VectorQuantizer_6786048328309
// MI455X (gfx1250) — matrix operand order repaired and checked by reading, not compiled, not run on hardware
//
#include <hip/hip_runtime.h>

#define NUM_EMB 4096
#define EMB_DIM 256
#define BM 128                 // rows per workgroup (8 waves x 16 rows)
#define BSTR 264               // padded f16 row stride in LDS B tile (conflict-free)
#define NTILES (NUM_EMB / 16)  // 256 code tiles
#define KSTEPS (EMB_DIM / 32)  // 8 k-steps of 32

typedef __attribute__((ext_vector_type(16))) _Float16 v16h;
typedef __attribute__((ext_vector_type(8)))  _Float16 v8h;
typedef __attribute__((ext_vector_type(8)))  float    v8f;
typedef __attribute__((ext_vector_type(4)))  int      v4i;

#if defined(__has_builtin)
#if __has_builtin(__builtin_amdgcn_global_load_async_to_lds_b128) && \
    __has_builtin(__builtin_amdgcn_s_wait_asynccnt)
#define HAVE_ASYNC_LDS 1
#endif
#endif
#ifndef HAVE_ASYNC_LDS
#define HAVE_ASYNC_LDS 0
#endif

#if HAVE_ASYNC_LDS
typedef __attribute__((address_space(1))) v4i as1_v4i;  // global
typedef __attribute__((address_space(3))) v4i as3_v4i;  // LDS
__device__ __forceinline__ void async_cp16(const void* g, void* l) {
  // GLOBAL_LOAD_ASYNC_TO_LDS_B128: memory -> LDS, tracked by ASYNCcnt
  __builtin_amdgcn_global_load_async_to_lds_b128((as1_v4i*)g, (as3_v4i*)l, 0, 0);
}
#endif

// ---------------- Kernel 1: codebook hi/lo f16 split + 0.5*||e||^2 ----------------
__global__ void vq_prep(const float* __restrict__ cb,
                        _Float16* __restrict__ cb_hi,
                        _Float16* __restrict__ cb_lo,
                        float* __restrict__ half_norm) {
  __shared__ float red[256];
  const int k = blockIdx.x;
  const int t = threadIdx.x;
  const float v = cb[(size_t)k * EMB_DIM + t];
  const _Float16 hi = (_Float16)v;
  const _Float16 lo = (_Float16)(v - (float)hi);
  cb_hi[(size_t)k * EMB_DIM + t] = hi;
  cb_lo[(size_t)k * EMB_DIM + t] = lo;
  red[t] = v * v;
  __syncthreads();
  for (int s = 128; s > 0; s >>= 1) {
    if (t < s) red[t] += red[t + s];
    __syncthreads();
  }
  if (t == 0) half_norm[k] = 0.5f * red[0];
}

// ---------------- Kernel 2: fused GEMM + argmax via WMMA ----------------
__global__ __launch_bounds__(256) void vq_argmin(
    const float* __restrict__ x,
    const _Float16* __restrict__ cb_hi,
    const _Float16* __restrict__ cb_lo,
    const float* __restrict__ half_norm,
    int* __restrict__ best_idx) {
  __shared__ __align__(16) _Float16 Bt[2][2][16 * BSTR];  // [buf][hi/lo][code*BSTR+d]
  __shared__ float hnl[NUM_EMB];                          // all half-norms, read per tile

  const int tid   = threadIdx.x;
  const int lane  = tid & 31;
  const int wave  = tid >> 5;
  const int halfw = lane >> 4;   // 0 or 1 (half-wave)
  const int l15   = lane & 15;

  // stage half-norms into LDS once (visible after the first barrier below)
  for (int i = tid; i < NUM_EMB; i += 256) hnl[i] = half_norm[i];

  // ---- Load + split this wave's A block: 16 rows x 256 (ISA 16x32 f16 A layout)
  // lane<16: row=l15, chunks K[0..7] & K[16..23]; lane>=16: K[8..15] & K[24..31]
  const int row = blockIdx.x * BM + wave * 16 + l15;
  const float* xr = x + (size_t)row * EMB_DIM;

  v16h a_hi[KSTEPS], a_lo[KSTEPS];
#pragma unroll
  for (int s = 0; s < KSTEPS; ++s) {
    const int d0 = s * 32 + halfw * 8;
    float4 f0 = *(const float4*)(xr + d0);
    float4 f1 = *(const float4*)(xr + d0 + 4);
    float4 f2 = *(const float4*)(xr + d0 + 16);
    float4 f3 = *(const float4*)(xr + d0 + 20);
    float va[16] = {f0.x, f0.y, f0.z, f0.w, f1.x, f1.y, f1.z, f1.w,
                    f2.x, f2.y, f2.z, f2.w, f3.x, f3.y, f3.z, f3.w};
    v16h hi, lo;
#pragma unroll
    for (int i = 0; i < 16; ++i) {
      _Float16 h = (_Float16)va[i];
      hi[i] = h;
      lo[i] = (_Float16)(va[i] - (float)h);
    }
    a_hi[s] = hi;
    a_lo[s] = lo;
  }

  float bestv[8];
  int   besti[8];
#pragma unroll
  for (int r = 0; r < 8; ++r) { bestv[r] = -3.0e38f; besti[r] = 0; }

  const int scode  = tid >> 4;   // code within tile this thread stages
  const int schunk = tid & 15;   // 16-f16 chunk within the code row

  // three independent accumulator chains -> 3-way WMMA ILP (no serial C-chain)
  auto computeTile = [&](int tile, int buf) {
    const int n = tile * 16 + l15;
    const float nb = -hnl[n];
    v8f acc0, acc1, acc2;
#pragma unroll
    for (int i = 0; i < 8; ++i) { acc0[i] = nb; acc1[i] = 0.0f; acc2[i] = 0.0f; }

    const _Float16* bh_base = &Bt[buf][0][l15 * BSTR + halfw * 8];
    const _Float16* bl_base = &Bt[buf][1][l15 * BSTR + halfw * 8];
#pragma unroll
    for (int s = 0; s < KSTEPS; ++s) {
      v8h bh0 = *(const v8h*)(bh_base + s * 32);
      v8h bh1 = *(const v8h*)(bh_base + s * 32 + 16);
      v8h bl0 = *(const v8h*)(bl_base + s * 32);
      v8h bl1 = *(const v8h*)(bl_base + s * 32 + 16);
      v16h bh = __builtin_shufflevector(bh0, bh1, 0,1,2,3,4,5,6,7,8,9,10,11,12,13,14,15);
      v16h bl = __builtin_shufflevector(bl0, bl1, 0,1,2,3,4,5,6,7,8,9,10,11,12,13,14,15);
      // split-precision product: xh*eh + xh*el + xl*eh (each its own chain)
      acc0 = __builtin_amdgcn_wmma_f32_16x16x32_f16(false, a_hi[s], false, bh, (short)0, acc0, false, false);
      acc1 = __builtin_amdgcn_wmma_f32_16x16x32_f16(false, a_hi[s], false, bl, (short)0, acc1, false, false);
      acc2 = __builtin_amdgcn_wmma_f32_16x16x32_f16(false, a_lo[s], false, bh, (short)0, acc2, false, false);
    }
#pragma unroll
    for (int r = 0; r < 8; ++r) {
      const float sc = acc0[r] + (acc1[r] + acc2[r]);
      if (sc > bestv[r]) { bestv[r] = sc; besti[r] = n; }
    }
  };

#if HAVE_ASYNC_LDS
  // -------- async global->LDS staging (ASYNCcnt), no VGPR round-trip --------
  auto issueTile = [&](int t, int b) {
    const _Float16* gh = cb_hi + (size_t)(t * 16 + scode) * EMB_DIM + schunk * 16;
    const _Float16* gl = cb_lo + (size_t)(t * 16 + scode) * EMB_DIM + schunk * 16;
    _Float16* dh = &Bt[b][0][scode * BSTR + schunk * 16];
    _Float16* dl = &Bt[b][1][scode * BSTR + schunk * 16];
    async_cp16(gh, dh);
    async_cp16(gh + 8, dh + 8);
    async_cp16(gl, dl);
    async_cp16(gl + 8, dl + 8);
  };
  issueTile(0, 0);
  __builtin_amdgcn_s_wait_asynccnt(0);
  __syncthreads();
  for (int tile = 0; tile < NTILES; ++tile) {
    const int buf = tile & 1;
    if (tile + 1 < NTILES) issueTile(tile + 1, buf ^ 1);  // overlap with compute
    computeTile(tile, buf);
    __builtin_amdgcn_s_wait_asynccnt(0);
    __syncthreads();
  }
#else
  // -------- fallback: global->reg->LDS double-buffered staging --------
  uint4 rh0, rh1, rl0, rl1;
  {
    const uint4* gh = (const uint4*)(cb_hi + (size_t)scode * EMB_DIM + schunk * 16);
    const uint4* gl = (const uint4*)(cb_lo + (size_t)scode * EMB_DIM + schunk * 16);
    rh0 = gh[0]; rh1 = gh[1]; rl0 = gl[0]; rl1 = gl[1];
  }
  for (int tile = 0; tile < NTILES; ++tile) {
    const int buf = tile & 1;
    {
      uint4* dh = (uint4*)&Bt[buf][0][scode * BSTR + schunk * 16];
      uint4* dl = (uint4*)&Bt[buf][1][scode * BSTR + schunk * 16];
      dh[0] = rh0; dh[1] = rh1; dl[0] = rl0; dl[1] = rl1;
    }
    __syncthreads();
    if (tile + 1 < NTILES) {
      const uint4* gh = (const uint4*)(cb_hi + (size_t)((tile + 1) * 16 + scode) * EMB_DIM + schunk * 16);
      const uint4* gl = (const uint4*)(cb_lo + (size_t)((tile + 1) * 16 + scode) * EMB_DIM + schunk * 16);
      rh0 = gh[0]; rh1 = gh[1]; rl0 = gl[0]; rl1 = gl[1];
    }
    computeTile(tile, buf);
  }
#endif

  // ---- reduce across the 16 column-classes (xor butterfly stays within each half-wave)
#pragma unroll
  for (int m = 1; m <= 8; m <<= 1) {
#pragma unroll
    for (int r = 0; r < 8; ++r) {
      float ov = __shfl_xor(bestv[r], m, 32);
      int   oi = __shfl_xor(besti[r], m, 32);
      if (ov > bestv[r] || (ov == bestv[r] && oi < besti[r])) {
        bestv[r] = ov; besti[r] = oi;
      }
    }
  }
  if (l15 == 0) {
    // C-layout: VGPR r, lane L -> M = r + 8*(L>=16)
    const int mbase = blockIdx.x * BM + wave * 16 + halfw * 8;
#pragma unroll
    for (int r = 0; r < 8; ++r) best_idx[mbase + r] = besti[r];
  }
}

// ---------------- Kernel 3: gather + straight-through output + SSE partials ----------------
__global__ void vq_gather(const float* __restrict__ x,
                          const float* __restrict__ cb,
                          const int* __restrict__ best_idx,
                          float* __restrict__ out,
                          float* __restrict__ partial) {
  __shared__ float red[256];
  const int t   = threadIdx.x;
  const int row = blockIdx.x * 4 + (t >> 6);
  const int d   = (t & 63) * 4;
  const int k   = best_idx[row];
  float4 q  = *(const float4*)(cb + (size_t)k * EMB_DIM + d);
  float4 xv = *(const float4*)(x + (size_t)row * EMB_DIM + d);
  float4 dv, o;
  dv.x = q.x - xv.x; dv.y = q.y - xv.y; dv.z = q.z - xv.z; dv.w = q.w - xv.w;
  // mimic reference: x + stop_grad(q - x)
  o.x = xv.x + dv.x; o.y = xv.y + dv.y; o.z = xv.z + dv.z; o.w = xv.w + dv.w;
  *(float4*)(out + (size_t)row * EMB_DIM + d) = o;
  red[t] = dv.x * dv.x + dv.y * dv.y + dv.z * dv.z + dv.w * dv.w;
  __syncthreads();
  for (int s = 128; s > 0; s >>= 1) {
    if (t < s) red[t] += red[t + s];
    __syncthreads();
  }
  if (t == 0) partial[blockIdx.x] = red[0];
}

// ---------------- Kernel 4: deterministic loss reduction ----------------
__global__ void vq_loss(const float* __restrict__ partial, int nb, float scale,
                        float* __restrict__ loss_out) {
  __shared__ float red[256];
  const int t = threadIdx.x;
  float s = 0.0f;
  for (int i = t; i < nb; i += 256) s += partial[i];
  red[t] = s;
  __syncthreads();
  for (int k = 128; k > 0; k >>= 1) {
    if (t < k) red[t] += red[t + k];
    __syncthreads();
  }
  if (t == 0) *loss_out = red[0] * scale;
}

extern "C" void kernel_launch(void* const* d_in, const int* in_sizes, int n_in,
                              void* d_out, int out_size, void* d_ws, size_t ws_size,
                              hipStream_t stream) {
  const float* x  = (const float*)d_in[0];
  const float* cb = (const float*)d_in[1];
  const int N = in_sizes[0] / EMB_DIM;  // 65536
  float* out = (float*)d_out;

  char* ws = (char*)d_ws;
  _Float16* cb_hi    = (_Float16*)ws;                              // 2 MB
  _Float16* cb_lo    = (_Float16*)(ws + (size_t)2 * 1024 * 1024);  // 2 MB
  float*    half_nrm = (float*)(ws + (size_t)4 * 1024 * 1024);     // 16 KB
  int*      best_idx = (int*)(ws + (size_t)4 * 1024 * 1024 + 16 * 1024);
  float*    partial  = (float*)(ws + (size_t)4 * 1024 * 1024 + 16 * 1024 + (size_t)N * sizeof(int));

  vq_prep<<<NUM_EMB, 256, 0, stream>>>(cb, cb_hi, cb_lo, half_nrm);
  vq_argmin<<<N / BM, 256, 0, stream>>>(x, cb_hi, cb_lo, half_nrm, best_idx);
  const int gblocks = N / 4;
  vq_gather<<<gblocks, 256, 0, stream>>>(x, cb, best_idx, out, partial);
  vq_loss<<<1, 256, 0, stream>>>(partial, gblocks, 1.25f / ((float)N * (float)EMB_DIM),
                                 out + (size_t)N * EMB_DIM);
}
